// SelfAttention_87119116632379
// MI455X (gfx1250) — compile-verified, matrix operand order checked by reading, not run on hardware
//
#include <hip/hip_runtime.h>
#include <hip/hip_bf16.h>
#include <cstdint>

// ---------------------------------------------------------------------------
// Self-attention over the batch dim (n = S*H problems of seqlen B=128, dh=64)
// All GEMMs via v_wmma_f32_16x16x32_bf16 (fp32 accumulate).
// ---------------------------------------------------------------------------

typedef __bf16 bf16;
typedef __attribute__((ext_vector_type(16))) __bf16 bf16x16;
typedef __attribute__((ext_vector_type(8)))  __bf16 bf16x8;
typedef __attribute__((ext_vector_type(8)))  float  f32x8;
typedef __attribute__((ext_vector_type(4)))  float  f32x4;
typedef __attribute__((ext_vector_type(4)))  int    i32x4;

union BF16x16 { bf16x16 v; bf16x8 h[2]; };

#define WMMA_BF16(A, B, C) \
  __builtin_amdgcn_wmma_f32_16x16x32_bf16(false, (A), false, (B), (short)0, (C), false, false)

// gfx1250 async global->LDS path (ASYNCcnt), with safe fallback.
#if defined(__AMDGCN__) && \
    __has_builtin(__builtin_amdgcn_global_load_async_to_lds_b128) && \
    __has_builtin(__builtin_amdgcn_s_wait_asynccnt)
#define HAVE_ASYNC_LDS 1
typedef __attribute__((address_space(1))) i32x4 as1_i32x4;   // global b128
typedef __attribute__((address_space(3))) i32x4 as3_i32x4;   // LDS b128
#else
#define HAVE_ASYNC_LDS 0
#endif

// Problem constants
#define BB 128
#define SS 512
#define DD 768
#define HH 12
#define DH 64
#define E3 2304   // 3*D

// Load one 16-bf16 A/B fragment (two 16B chunks) from a row pointer.
__device__ __forceinline__ void load_frag(BF16x16& f, const bf16* row,
                                          int k0, int half) {
  f.h[0] = *(const bf16x8*)(row + k0 + half * 8);
  f.h[1] = *(const bf16x8*)(row + k0 + 16 + half * 8);
}

// ---------------------------------------------------------------------------
// Kernel 0: convert weights fp32 -> bf16 in workspace; copy b_dense to d_out.
// ---------------------------------------------------------------------------
__global__ __launch_bounds__(256) void sa_prep_kernel(
    const float* __restrict__ Wq, const float* __restrict__ Wd,
    const float* __restrict__ bd,
    bf16* __restrict__ wq_b, bf16* __restrict__ wd_b, float* __restrict__ out_tail)
{
  const size_t NQ = (size_t)E3 * DD;   // 1,769,472
  const size_t ND = (size_t)DD * DD;   //   589,824
  size_t i = (size_t)blockIdx.x * 256 + threadIdx.x;
  if (i < NQ)  wq_b[i] = (bf16)Wq[i];
  if (i < ND)  wd_b[i] = (bf16)Wd[i];
  if (i < DD)  out_tail[i] = bd[i];
}

// ---------------------------------------------------------------------------
// Kernel 1: X fp32 -> bf16, one pass (8 elements / thread, b128 in, b128 out)
// ---------------------------------------------------------------------------
__global__ __launch_bounds__(256) void sa_xconv_kernel(
    const float* __restrict__ X, bf16* __restrict__ Xb)
{
  size_t i = ((size_t)blockIdx.x * 256 + threadIdx.x) * 8;
  f32x4 a0 = *(const f32x4*)(X + i);
  f32x4 a1 = *(const f32x4*)(X + i + 4);
  bf16x8 o;
  #pragma unroll
  for (int j = 0; j < 4; ++j) { o[j] = (bf16)a0[j]; o[4 + j] = (bf16)a1[j]; }
  *(bf16x8*)(Xb + i) = o;
}

// ---------------------------------------------------------------------------
// Kernel 2: qkv = Xb * W_qkv^T + b_qkv (bf16 in, bf16 out), written into
//   q_buf[(s*H+h)][b][dh], k_buf[(s*H+h)][b][dh], v_buf[(s*H+h)][dh][b]
// Block = 8 waves = 4 m-tiles x 2 e-groups; each wave computes a 16x64 strip
// (4 accumulators). K-loop is register double-buffered so each WMMA group
// depends only on the previous iteration's loads (non-zero loadcnt waits).
// ---------------------------------------------------------------------------
__global__ __launch_bounds__(256) void sa_qkv_kernel(
    const bf16* __restrict__ Xb, const bf16* __restrict__ Wq,
    const float* __restrict__ bqkv,
    bf16* __restrict__ q_buf, bf16* __restrict__ k_buf, bf16* __restrict__ v_buf)
{
  const int lane = threadIdx.x & 31;
  const int wave = threadIdx.x >> 5;
  const int half = lane >> 4;
  const int lm   = lane & 15;

  const int eg = blockIdx.x * 2 + (wave >> 2);   // 0..35  (64-col e-group)
  const int mt = blockIdx.y * 4 + (wave & 3);    // 0..7   (batch tile)
  const int s  = blockIdx.z;                     // 0..511
  const int e0 = eg * 64;
  const int b0 = mt * 16;

  const bf16* arow = Xb + ((size_t)(b0 + lm) * SS + s) * DD;
  const bf16* brow = Wq + (size_t)(e0 + lm) * DD;

  f32x8 acc[4];
  #pragma unroll
  for (int t = 0; t < 4; ++t) acc[t] = (f32x8){};

  // -------- software-pipelined K loop (double buffered fragments) ----------
  BF16x16 ca, cb[4];
  load_frag(ca, arow, 0, half);
  #pragma unroll
  for (int t = 0; t < 4; ++t) load_frag(cb[t], brow + (size_t)t * 16 * DD, 0, half);

  for (int k0 = 32; k0 < DD; k0 += 32) {
    BF16x16 na, nb[4];
    load_frag(na, arow, k0, half);
    #pragma unroll
    for (int t = 0; t < 4; ++t) load_frag(nb[t], brow + (size_t)t * 16 * DD, k0, half);
    __builtin_prefetch(arow + k0 + 224, 0, 0);   // global_prefetch_b8

    #pragma unroll
    for (int t = 0; t < 4; ++t) acc[t] = WMMA_BF16(ca.v, cb[t].v, acc[t]);

    ca = na;
    #pragma unroll
    for (int t = 0; t < 4; ++t) cb[t] = nb[t];
  }
  #pragma unroll
  for (int t = 0; t < 4; ++t) acc[t] = WMMA_BF16(ca.v, cb[t].v, acc[t]);

  // D layout: lane column n = t*16 + lm, register r -> row m = r + 8*half
  const int h     = eg / 3;
  const int which = eg % 3;      // 0=q, 1=k, 2=v  (dh0 == 0 by construction)
  const size_t nprob = (size_t)s * HH + h;

  if (which == 2) {
    bf16* dst = v_buf + nprob * (size_t)(DH * BB);   // [dh][b]
    #pragma unroll
    for (int t = 0; t < 4; ++t) {
      const float bias = bqkv[e0 + t * 16 + lm];
      #pragma unroll
      for (int r = 0; r < 8; ++r) {
        const int b = b0 + r + 8 * half;
        dst[(size_t)(t * 16 + lm) * BB + b] = (bf16)(acc[t][r] + bias);
      }
    }
  } else {
    bf16* dst = (which == 0 ? q_buf : k_buf) + nprob * (size_t)(BB * DH); // [b][dh]
    #pragma unroll
    for (int t = 0; t < 4; ++t) {
      const float bias = bqkv[e0 + t * 16 + lm];
      #pragma unroll
      for (int r = 0; r < 8; ++r) {
        const int b = b0 + r + 8 * half;
        dst[(size_t)b * DH + t * 16 + lm] = (bf16)(acc[t][r] + bias);
      }
    }
  }
}

// ---------------------------------------------------------------------------
// Kernel 3: per-(s,h) attention. scores = Q K^T / 8, mask where j<=i to -1e4,
// softmax rows, ctx = P V. One block per problem; 8 waves x 16 query rows.
// K/V staged into LDS via gfx1250 async global->LDS (ASYNCcnt) when available.
// P staged per-wave in LDS (bf16) to convert the D-layout to the A-layout.
// ---------------------------------------------------------------------------
__global__ __launch_bounds__(256) void sa_attn_kernel(
    const bf16* __restrict__ q_buf, const bf16* __restrict__ k_buf,
    const bf16* __restrict__ v_buf, bf16* __restrict__ ctx)
{
  __shared__ __align__(16) bf16 lds_k[BB][DH];       // 16 KB
  __shared__ __align__(16) bf16 lds_v[DH][BB];       // 16 KB (transposed V)
  __shared__ __align__(16) bf16 lds_p[8][16][BB];    // 32 KB

  const int tid  = threadIdx.x;
  const int lane = tid & 31;
  const int wave = tid >> 5;
  const int half = lane >> 4;
  const int lm   = lane & 15;

  const int n = blockIdx.x;          // 0..6143
  const int s = n / HH;
  const int h = n % HH;

  const bf16* Q = q_buf + (size_t)n * (BB * DH);
  const bf16* K = k_buf + (size_t)n * (BB * DH);
  const bf16* V = v_buf + (size_t)n * (DH * BB);    // already [dh][b]

  // ---- stage K and V into LDS (4 x 16B each per thread) --------------------
  {
    bf16* lk = &lds_k[0][0];
    bf16* lv = &lds_v[0][0];
#if HAVE_ASYNC_LDS
    #pragma unroll
    for (int i = 0; i < 4; ++i) {
      const int off = (tid + i * 256) * 8;          // 0..8184 (elements)
      __builtin_amdgcn_global_load_async_to_lds_b128(
          (as1_i32x4*)(uintptr_t)(K + off),
          (as3_i32x4*)(uintptr_t)(uint32_t)(uintptr_t)(lk + off), 0, 0);
      __builtin_amdgcn_global_load_async_to_lds_b128(
          (as1_i32x4*)(uintptr_t)(V + off),
          (as3_i32x4*)(uintptr_t)(uint32_t)(uintptr_t)(lv + off), 0, 0);
    }
    __builtin_amdgcn_s_wait_asynccnt(0);
#else
    #pragma unroll
    for (int i = 0; i < 4; ++i) {
      const int off = (tid + i * 256) * 8;
      *(bf16x8*)(lk + off) = *(const bf16x8*)(K + off);
      *(bf16x8*)(lv + off) = *(const bf16x8*)(V + off);
    }
#endif
  }
  __syncthreads();

  const int m0 = wave * 16;          // query-row stripe

  // ---- scores: 16 x 128 stripe = 8 accumulator tiles -----------------------
  f32x8 acc[8];
  #pragma unroll
  for (int ct = 0; ct < 8; ++ct) acc[ct] = (f32x8){};

  #pragma unroll
  for (int k0 = 0; k0 < DH; k0 += 32) {
    BF16x16 a;
    load_frag(a, Q + (size_t)(m0 + lm) * DH, k0, half);
    #pragma unroll
    for (int ct = 0; ct < 8; ++ct) {
      BF16x16 bm;
      load_frag(bm, &lds_k[ct * 16 + lm][0], k0, half);
      acc[ct] = WMMA_BF16(a.v, bm.v, acc[ct]);
    }
  }

  // ---- scale + inverted-tril mask + row softmax -----------------------------
  // row i = m0 + r + 8*half ; col j = ct*16 + lm
  #pragma unroll
  for (int r = 0; r < 8; ++r) {
    const int i = m0 + r + 8 * half;
    float mx = -3.0e38f;
    #pragma unroll
    for (int ct = 0; ct < 8; ++ct) {
      const int j = ct * 16 + lm;
      float v = acc[ct][r] * 0.125f;          // 1/sqrt(64)
      if (j <= i) v = -10000.0f;              // reference masks the LOWER tri
      acc[ct][r] = v;
      mx = fmaxf(mx, v);
    }
    #pragma unroll
    for (int off = 1; off < 16; off <<= 1)
      mx = fmaxf(mx, __shfl_xor(mx, off, 32));   // stays within 16-lane half

    float sum = 0.0f;
    #pragma unroll
    for (int ct = 0; ct < 8; ++ct) {
      const float e = __expf(acc[ct][r] - mx);
      acc[ct][r] = e;
      sum += e;
    }
    #pragma unroll
    for (int off = 1; off < 16; off <<= 1)
      sum += __shfl_xor(sum, off, 32);

    const float inv = 1.0f / sum;
    #pragma unroll
    for (int ct = 0; ct < 8; ++ct)
      lds_p[wave][r + 8 * half][ct * 16 + lm] = (bf16)(acc[ct][r] * inv);
  }
  // Each wave reads back only its own lds_p stripe -> DScnt ordering suffices.

  // ---- ctx = P * V : 16 x 64 stripe = 4 accumulator tiles -------------------
  f32x8 o[4];
  #pragma unroll
  for (int nt = 0; nt < 4; ++nt) o[nt] = (f32x8){};

  #pragma unroll
  for (int c0 = 0; c0 < BB; c0 += 32) {
    BF16x16 a;
    load_frag(a, &lds_p[wave][lm][0], c0, half);
    #pragma unroll
    for (int nt = 0; nt < 4; ++nt) {
      BF16x16 bm;
      load_frag(bm, &lds_v[nt * 16 + lm][0], c0, half);
      o[nt] = WMMA_BF16(a.v, bm.v, o[nt]);
    }
  }

  // ---- store ctx in [b][s][d] order (bf16) ---------------------------------
  #pragma unroll
  for (int nt = 0; nt < 4; ++nt) {
    #pragma unroll
    for (int r = 0; r < 8; ++r) {
      const int i = m0 + r + 8 * half;   // query batch index
      ctx[((size_t)i * SS + s) * DD + h * DH + nt * 16 + lm] = (bf16)o[nt][r];
    }
  }
}

// ---------------------------------------------------------------------------
// Kernel 4: out = ctx * W_dense^T  (fp32 output, no bias — returned separately)
// Block = 8 waves = 4 m-tiles x 2 e-groups; wave computes a 16x64 strip.
// Same register double-buffering as the QKV GEMM.
// ---------------------------------------------------------------------------
__global__ __launch_bounds__(256) void sa_dense_kernel(
    const bf16* __restrict__ ctx, const bf16* __restrict__ Wd,
    float* __restrict__ out)
{
  const int lane = threadIdx.x & 31;
  const int wave = threadIdx.x >> 5;
  const int half = lane >> 4;
  const int lm   = lane & 15;

  const int eg = blockIdx.x * 2 + (wave >> 2);   // 0..11
  const int mt = blockIdx.y * 4 + (wave & 3);    // 0..4095
  const int e0 = eg * 64;
  const int m0 = mt * 16;

  const bf16* arow = ctx + (size_t)(m0 + lm) * DD;
  const bf16* brow = Wd  + (size_t)(e0 + lm) * DD;

  f32x8 acc[4];
  #pragma unroll
  for (int t = 0; t < 4; ++t) acc[t] = (f32x8){};

  BF16x16 ca, cb[4];
  load_frag(ca, arow, 0, half);
  #pragma unroll
  for (int t = 0; t < 4; ++t) load_frag(cb[t], brow + (size_t)t * 16 * DD, 0, half);

  for (int k0 = 32; k0 < DD; k0 += 32) {
    BF16x16 na, nb[4];
    load_frag(na, arow, k0, half);
    #pragma unroll
    for (int t = 0; t < 4; ++t) load_frag(nb[t], brow + (size_t)t * 16 * DD, k0, half);
    __builtin_prefetch(arow + k0 + 224, 0, 0);

    #pragma unroll
    for (int t = 0; t < 4; ++t) acc[t] = WMMA_BF16(ca.v, cb[t].v, acc[t]);

    ca = na;
    #pragma unroll
    for (int t = 0; t < 4; ++t) cb[t] = nb[t];
  }
  #pragma unroll
  for (int t = 0; t < 4; ++t) acc[t] = WMMA_BF16(ca.v, cb[t].v, acc[t]);

  #pragma unroll
  for (int t = 0; t < 4; ++t)
    #pragma unroll
    for (int r = 0; r < 8; ++r)
      out[(size_t)(m0 + r + 8 * half) * DD + e0 + t * 16 + lm] = acc[t][r];
}

// ---------------------------------------------------------------------------
// Launch
// ---------------------------------------------------------------------------
extern "C" void kernel_launch(void* const* d_in, const int* in_sizes, int n_in,
                              void* d_out, int out_size, void* d_ws, size_t ws_size,
                              hipStream_t stream)
{
  (void)in_sizes; (void)n_in; (void)out_size; (void)ws_size;

  const float* X  = (const float*)d_in[0];   // [128,512,768]
  const float* Wq = (const float*)d_in[1];   // [2304,768]
  const float* bq = (const float*)d_in[2];   // [2304]
  const float* Wd = (const float*)d_in[3];   // [768,768]
  const float* bd = (const float*)d_in[4];   // [768]
  float* out = (float*)d_out;                // 65536*768 f32, then 768 f32 bias

  // Workspace layout (bytes), all 256-aligned. Total ~389 MB.
  // Xb (bf16 X) is dead after the QKV GEMM, so ctx aliases it.
  char* ws = (char*)d_ws;
  bf16* wq_b  = (bf16*)(ws + 0);            //  3,538,944 B
  bf16* wd_b  = (bf16*)(ws + 3538944);      //  1,179,648 B
  bf16* xb    = (bf16*)(ws + 4718592);      // 100,663,296 B  [128*512][768]
  bf16* ctx   = xb;                         // alias (attn runs after qkv)
  bf16* q_buf = (bf16*)(ws + 105381888);    // 100,663,296 B  [6144][128][64]
  bf16* k_buf = (bf16*)(ws + 206045184);    // 100,663,296 B  [6144][128][64]
  bf16* v_buf = (bf16*)(ws + 306708480);    // 100,663,296 B  [6144][64][128]

  sa_prep_kernel<<<6912, 256, 0, stream>>>(Wq, Wd, bd, wq_b, wd_b,
                                           out + (size_t)65536 * 768);
  sa_xconv_kernel<<<24576, 256, 0, stream>>>(X, xb);
  sa_qkv_kernel<<<dim3(18, 2, 512), 256, 0, stream>>>(xb, wq_b, bq,
                                                      q_buf, k_buf, v_buf);
  sa_attn_kernel<<<6144, 256, 0, stream>>>(q_buf, k_buf, v_buf, ctx);
  sa_dense_kernel<<<dim3(6, 1024), 256, 0, stream>>>(ctx, wd_b, out);
}
